// ESTLayer_15436112462036
// MI455X (gfx1250) — compile-verified, matrix operand order checked by reading, not run on hardware
//
#include <hip/hip_runtime.h>
#include <hip/hip_bf16.h>
#include <math.h>

// Problem sizes (fixed by reference)
#define B_SZ 256
#define U_SZ 8
#define N_SZ 2048
#define I_SZ 1024
#define O_SZ 1024

// Tiling
#define BM 128      // batch-rows per block
#define BN 128      // output-cols per block
#define BK 32       // K per step (one bf16 WMMA K)
#define LDS_K 40    // padded K stride in elements (80B rows: 16B-aligned, staggered banks)

typedef __attribute__((ext_vector_type(16))) __bf16 v16bf;
typedef __attribute__((ext_vector_type(4)))  __bf16 v4bf;
typedef __attribute__((ext_vector_type(8)))  float  v8f;
typedef __attribute__((ext_vector_type(4)))  float  v4f;

static __device__ __forceinline__ __bf16 f2bf(float f) { return (__bf16)f; }

// Per-thread register staging for one (A 1x16, B 4x4) slice of the K-step tile.
struct TileRegs {
    v4f a[4];   // 16 consecutive floats of one A row
    v4f b[4];   // 4 consecutive n-floats from 4 consecutive k-rows of B
};

static __device__ __forceinline__ void load_tile(
    TileRegs& r, const float* __restrict__ aPtr,
    const float* __restrict__ bPtr, int strideB, int g, int nch)
{
    #pragma unroll
    for (int i = 0; i < 4; ++i) r.a[i] = *(const v4f*)(aPtr + i * 4);
    #pragma unroll
    for (int i = 0; i < 4; ++i)
        r.b[i] = *(const v4f*)(bPtr + (size_t)(g * 4 + i) * strideB + nch);
}

static __device__ __forceinline__ void store_tile(
    const TileRegs& r, __bf16* sA, __bf16* sB,
    int row, int h, int g, int nch, float scl)
{
    // A: 16 bf16 contiguous -> 32B LDS store (2x ds_store_b128)
    v16bf av;
    #pragma unroll
    for (int i = 0; i < 4; ++i)
        #pragma unroll
        for (int j = 0; j < 4; ++j)
            av[i * 4 + j] = f2bf(r.a[i][j]);
    *(v16bf*)&sA[row * LDS_K + h * 16] = av;

    // B: 4x4 register transpose -> 4x ds_store_b64 into [n][k] layout
    #pragma unroll
    for (int j = 0; j < 4; ++j) {
        v4bf c;
        c[0] = f2bf(r.b[0][j] * scl);
        c[1] = f2bf(r.b[1][j] * scl);
        c[2] = f2bf(r.b[2][j] * scl);
        c[3] = f2bf(r.b[3][j] * scl);
        *(v4bf*)&sB[(nch + j) * LDS_K + g * 4] = c;
    }
}

static __device__ __forceinline__ void compute_step(
    const __bf16* sA, const __bf16* sB, int wm, int wn, int lane, v8f acc[4][2])
{
    const int khalf = (lane >> 4) * 16;   // lanes 0-15: K 0..15, 16-31: K 16..31
    const int r16   = lane & 15;
    v16bf aF[4], bF[2];
    #pragma unroll
    for (int mt = 0; mt < 4; ++mt)
        aF[mt] = *(const v16bf*)&sA[(wm * 64 + mt * 16 + r16) * LDS_K + khalf];
    #pragma unroll
    for (int nt = 0; nt < 2; ++nt)
        bF[nt] = *(const v16bf*)&sB[(wn * 32 + nt * 16 + r16) * LDS_K + khalf];
    #pragma unroll
    for (int mt = 0; mt < 4; ++mt)
        #pragma unroll
        for (int nt = 0; nt < 2; ++nt)
            acc[mt][nt] = __builtin_amdgcn_wmma_f32_16x16x32_bf16(
                false, aF[mt], false, bF[nt], (short)0, acc[mt][nt],
                false, false);
}

// ---------------------------------------------------------------------------
// Kernel 1: lr[b,u] = softmax_u( dot(X[b,u,:], alr[u,:]) / temperature )
// ---------------------------------------------------------------------------
__global__ __launch_bounds__(256) void lr_softmax_kernel(
    const float* __restrict__ X, const float* __restrict__ alr,
    const float* __restrict__ temperature, float* __restrict__ lr_out)
{
    const int b    = blockIdx.x;
    const int tid  = threadIdx.x;
    const int u    = tid >> 5;
    const int lane = tid & 31;

    __shared__ float sdot[U_SZ];

    const float* __restrict__ xrow = X   + ((size_t)b * U_SZ + u) * I_SZ;
    const float* __restrict__ arow = alr + (size_t)u * I_SZ;

    float acc = 0.0f;
    #pragma unroll 8
    for (int j = lane; j < I_SZ; j += 32)
        acc = fmaf(xrow[j], arow[j], acc);

    #pragma unroll
    for (int off = 16; off > 0; off >>= 1)
        acc += __shfl_down(acc, off, 32);

    if (lane == 0) sdot[u] = acc / temperature[0];
    __syncthreads();

    if (tid < U_SZ) {
        float mx = sdot[0];
        #pragma unroll
        for (int i = 1; i < U_SZ; ++i) mx = fmaxf(mx, sdot[i]);
        float sum = 0.0f;
        #pragma unroll
        for (int i = 0; i < U_SZ; ++i) sum += __expf(sdot[i] - mx);
        lr_out[b * U_SZ + tid] = __expf(sdot[tid] - mx) / sum;
    }
}

// ---------------------------------------------------------------------------
// Kernel 2: fused pre-activation GEMM + state update (K = I + N = 3072).
//   pre = [X_u | state_u] @ [Win_u ; sr_u*W_u] + bias
//   new_state = (1-lr)*state + lr*tanh(pre)
// Double-buffered LDS; 8 waves (2x4), each wave 4x2 WMMA tiles.
// ---------------------------------------------------------------------------
__global__ __launch_bounds__(256) void fused_state_kernel(
    const float* __restrict__ X,    const float* __restrict__ state,
    const float* __restrict__ W,    const float* __restrict__ Win,
    const float* __restrict__ bias, const float* __restrict__ sr,
    const float* __restrict__ lr_buf, float* __restrict__ new_state)
{
    __shared__ __bf16 sA[2][BM * LDS_K];
    __shared__ __bf16 sB[2][BN * LDS_K];

    const int u    = blockIdx.z;
    const int n0   = blockIdx.x * BN;
    const int m0   = blockIdx.y * BM;
    const int tid  = threadIdx.x;
    const int lane = tid & 31;
    const int wid  = tid >> 5;
    const int wm   = wid >> 2;
    const int wn   = wid & 3;
    const float srs = sr[u];

    // staging assignments
    const int row = tid >> 1;           // A row 0..127
    const int h   = tid & 1;            // A half (16 floats)
    const int g   = tid >> 5;           // B k-group 0..7 (4 k-rows each)
    const int nch = (tid & 31) * 4;     // B n-chunk

    const float* __restrict__ Xrow = X     + ((size_t)(m0 + row) * U_SZ + u) * I_SZ + h * 16;
    const float* __restrict__ Srow = state + ((size_t)(m0 + row) * U_SZ + u) * N_SZ + h * 16;
    const float* __restrict__ WinB = Win + (size_t)u * I_SZ * N_SZ + n0;
    const float* __restrict__ WB   = W   + (size_t)u * N_SZ * N_SZ + n0;

    v8f zeroV = {};
    v8f acc[4][2];
    #pragma unroll
    for (int mt = 0; mt < 4; ++mt)
        #pragma unroll
        for (int nt = 0; nt < 2; ++nt) acc[mt][nt] = zeroV;

    const int KTOT = I_SZ + N_SZ;

    TileRegs regs;
    // prologue: stage K-step 0 (always in the X/Win segment)
    load_tile(regs, Xrow, WinB, N_SZ, g, nch);
    store_tile(regs, sA[0], sB[0], row, h, g, nch, 1.0f);

    int p = 0;
    for (int kk = 0; kk < KTOT; kk += BK) {
        __syncthreads();                      // buf p staged for everyone
        const int  kn        = kk + BK;
        const bool have_next = kn < KTOT;
        float sclN = 1.0f;
        if (have_next) {                      // issue next tile's global loads
            const float* aP;
            const float* bP;
            if (kn < I_SZ) { aP = Xrow + kn;          bP = WinB + (size_t)kn * N_SZ; }
            else { aP = Srow + (kn - I_SZ); bP = WB + (size_t)(kn - I_SZ) * N_SZ; sclN = srs; }
            load_tile(regs, aP, bP, N_SZ, g, nch);
        }
        compute_step(sA[p], sB[p], wm, wn, lane, acc);   // overlaps loads
        if (have_next)
            store_tile(regs, sA[1 - p], sB[1 - p], row, h, g, nch, sclN);
        p ^= 1;
    }

    // epilogue: bias + tanh + lr blend
    const int nloc = lane & 15;
    const int mhi  = (lane >> 4) * 8;   // C layout: VGPR v -> M=v (+8 for hi lanes)
    #pragma unroll
    for (int mt = 0; mt < 4; ++mt) {
        #pragma unroll
        for (int nt = 0; nt < 2; ++nt) {
            const int n   = n0 + wn * 32 + nt * 16 + nloc;
            const float bia = bias[u * N_SZ + n];
            #pragma unroll
            for (int v = 0; v < 8; ++v) {
                const int brow = m0 + wm * 64 + mt * 16 + mhi + v;
                const float pre = acc[mt][nt][v] + bia;
                const float t   = tanhf(pre);
                const float lr  = lr_buf[brow * U_SZ + u];
                const size_t idx = ((size_t)brow * U_SZ + u) * N_SZ + n;
                const float s = state[idx];
                new_state[idx] = (1.0f - lr) * s + lr * t;
            }
        }
    }
}

// ---------------------------------------------------------------------------
// Kernel 3: output = new_state_u (256 x 2048) @ Wout_u (2048 x 1024)
// ---------------------------------------------------------------------------
__global__ __launch_bounds__(256) void output_gemm_kernel(
    const float* __restrict__ new_state, const float* __restrict__ Wout,
    float* __restrict__ out)
{
    __shared__ __bf16 sA[2][BM * LDS_K];
    __shared__ __bf16 sB[2][BN * LDS_K];

    const int u    = blockIdx.z;
    const int n0   = blockIdx.x * BN;
    const int m0   = blockIdx.y * BM;
    const int tid  = threadIdx.x;
    const int lane = tid & 31;
    const int wid  = tid >> 5;
    const int wm   = wid >> 2;
    const int wn   = wid & 3;

    const int row = tid >> 1;
    const int h   = tid & 1;
    const int g   = tid >> 5;
    const int nch = (tid & 31) * 4;

    const float* __restrict__ Arow  = new_state + ((size_t)(m0 + row) * U_SZ + u) * N_SZ + h * 16;
    const float* __restrict__ WoutB = Wout + (size_t)u * N_SZ * O_SZ + n0;

    v8f zeroV = {};
    v8f acc[4][2];
    #pragma unroll
    for (int mt = 0; mt < 4; ++mt)
        #pragma unroll
        for (int nt = 0; nt < 2; ++nt) acc[mt][nt] = zeroV;

    TileRegs regs;
    load_tile(regs, Arow, WoutB, O_SZ, g, nch);
    store_tile(regs, sA[0], sB[0], row, h, g, nch, 1.0f);

    int p = 0;
    for (int kk = 0; kk < N_SZ; kk += BK) {
        __syncthreads();
        const int  kn        = kk + BK;
        const bool have_next = kn < N_SZ;
        if (have_next)
            load_tile(regs, Arow + kn, WoutB + (size_t)kn * O_SZ, O_SZ, g, nch);
        compute_step(sA[p], sB[p], wm, wn, lane, acc);
        if (have_next)
            store_tile(regs, sA[1 - p], sB[1 - p], row, h, g, nch, 1.0f);
        p ^= 1;
    }

    const int nloc = lane & 15;
    const int mhi  = (lane >> 4) * 8;
    #pragma unroll
    for (int mt = 0; mt < 4; ++mt) {
        #pragma unroll
        for (int nt = 0; nt < 2; ++nt) {
            const int o = n0 + wn * 32 + nt * 16 + nloc;
            #pragma unroll
            for (int v = 0; v < 8; ++v) {
                const int brow = m0 + wm * 64 + mt * 16 + mhi + v;
                out[((size_t)brow * U_SZ + u) * O_SZ + o] = acc[mt][nt][v];
            }
        }
    }
}

// ---------------------------------------------------------------------------
extern "C" void kernel_launch(void* const* d_in, const int* in_sizes, int n_in,
                              void* d_out, int out_size, void* d_ws, size_t ws_size,
                              hipStream_t stream) {
    const float* X     = (const float*)d_in[0];
    const float* state = (const float*)d_in[1];
    const float* W     = (const float*)d_in[2];
    const float* Win   = (const float*)d_in[3];
    const float* bias  = (const float*)d_in[4];
    const float* Wout  = (const float*)d_in[5];
    const float* alr   = (const float*)d_in[6];
    const float* sr    = (const float*)d_in[7];
    const float* temp  = (const float*)d_in[8];

    float* out_f     = (float*)d_out;
    float* new_state = out_f;                                 // (B,U,N)
    float* output    = out_f + (size_t)B_SZ * U_SZ * N_SZ;    // (B,U,O)
    float* lr_buf    = (float*)d_ws;                          // (B,U)

    lr_softmax_kernel<<<B_SZ, 256, 0, stream>>>(X, alr, temp, lr_buf);

    dim3 g2(N_SZ / BN, B_SZ / BM, U_SZ);
    fused_state_kernel<<<g2, 256, 0, stream>>>(X, state, W, Win, bias, sr,
                                               lr_buf, new_state);

    dim3 g3(O_SZ / BN, B_SZ / BM, U_SZ);
    output_gemm_kernel<<<g3, 256, 0, stream>>>(new_state, Wout, output);
}
